// SinCAA_65678639891118
// MI455X (gfx1250) — matrix operand order repaired and checked by reading, not compiled, not run on hardware
//
#include <hip/hip_runtime.h>
#include <hip/hip_bf16.h>

typedef unsigned short u16;
typedef __attribute__((ext_vector_type(16))) __bf16           v16bf;
typedef __attribute__((ext_vector_type(16))) unsigned short   v16u;
typedef __attribute__((ext_vector_type(8)))  unsigned short   v8u;
typedef __attribute__((ext_vector_type(8)))  float            v8f;

#define NN   131072          // nodes
#define CC   128             // channels
#define GG   1024            // graphs
#define NP_  128             // nodes per graph
#define EE   (1u<<20)        // edges
#define ECHUNK 131072        // edge-head chunk (EE/8)

__device__ __forceinline__ u16 f2bf(float f) {
    unsigned u = __float_as_uint(f);
    u += 0x7FFFu + ((u >> 16) & 1u);     // round-to-nearest-even
    return (u16)(u >> 16);
}

// ---------------------------------------------------------------------------
// WMMA GEMM: out[M,NP] = A[M,K](bf16) x Wt[NP,K](bf16)^T + bias (+prelu)(+residual)
// block 256 = 8 waves; each wave computes a 16(M) x 64(N) strip (4 accumulators,
// A fragment reused 4x); block tile 128(M) x 64(N).
// ---------------------------------------------------------------------------
__global__ void k_gemm(const u16* __restrict__ A, const u16* __restrict__ Bt,
                       const float* __restrict__ bias, const float* __restrict__ preluA,
                       const float* __restrict__ residual,
                       float* __restrict__ outF, u16* __restrict__ outB,
                       int M, int K, int NP, int NR) {
    const int tid  = threadIdx.x;
    const int lane = tid & 31;
    const int wave = tid >> 5;
    const int lm   = lane & 15;
    const int hi   = lane >> 4;                    // 0 or 1
    const int m0   = blockIdx.x * 128 + wave * 16;
    const int n0   = blockIdx.y * 64;

    // A: K slots {kb..kb+7} U {kb+16..kb+23}, kb = hi*8  -> two b128 loads
    const u16* ap = A  + (size_t)(m0 + lm) * K + hi * 8;
    // B: K slots {kb..kb+7} U {kb+16..kb+23}, kb = hi*8  -> two b128 loads
    const u16* bp = Bt + (size_t)(n0 + lm) * K + hi * 8;

    v8f acc[4];
#pragma unroll
    for (int t = 0; t < 4; ++t) acc[t] = (v8f){};

    const int steps = K >> 5;
    for (int kk = 0; kk < steps; ++kk) {
        v8u a0 = *reinterpret_cast<const v8u*>(ap);
        v8u a1 = *reinterpret_cast<const v8u*>(ap + 16);
        v16u av;
#pragma unroll
        for (int i = 0; i < 8; ++i) { av[i] = a0[i]; av[i + 8] = a1[i]; }
        const v16bf af = __builtin_bit_cast(v16bf, av);
#pragma unroll
        for (int t = 0; t < 4; ++t) {              // reuse A across 4 N-tiles
            const u16* bpt = bp + (size_t)t * 16 * K;
            v8u b0 = *reinterpret_cast<const v8u*>(bpt);
            v8u b1 = *reinterpret_cast<const v8u*>(bpt + 16);
            v16u bv;
#pragma unroll
            for (int i = 0; i < 8; ++i) { bv[i] = b0[i]; bv[i + 8] = b1[i]; }
            acc[t] = __builtin_amdgcn_wmma_f32_16x16x32_bf16(
                         false, af, false, __builtin_bit_cast(v16bf, bv),
                         (short)0, acc[t], false, false);
        }
        ap += 32; bp += 32;
    }

    const float pa = preluA ? *preluA : 0.f;
#pragma unroll
    for (int t = 0; t < 4; ++t) {
        const int n = n0 + t * 16 + lm;
        if (n < NR) {
            const float bn = bias ? bias[n] : 0.f;
#pragma unroll
            for (int r = 0; r < 8; ++r) {
                const int m = m0 + r + hi * 8;
                float v = acc[t][r] + bn;
                if (preluA)   v = (v >= 0.f) ? v : pa * v;
                if (residual) v += residual[(size_t)m * NP + n];
                const size_t o = (size_t)m * NP + n;
                if (outF) outF[o] = v;
                if (outB) outB[o] = f2bf(v);
            }
        }
    }
}

// weights: W[K,Nout] (f32) -> Wt[NP,K] (bf16), zero-padded rows
__global__ void k_prep_w(const float* __restrict__ W, u16* __restrict__ Bt,
                         int K, int Nout, int NP) {
    int idx = blockIdx.x * 256 + threadIdx.x;       // NP*K threads
    int n = idx / K, k = idx - n * K;
    float v = (n < Nout) ? W[(size_t)k * Nout + n] : 0.f;
    Bt[idx] = f2bf(v);
}

// x0 = node_float @ W4 + b + mean(int embeddings); then node dropout mask
__global__ void k_embed(const int* __restrict__ nif, const float* __restrict__ nff,
                        const float* __restrict__ mrand,
                        const float* __restrict__ W4, const float* __restrict__ b4,
                        const float* __restrict__ tEmb,
                        float* __restrict__ X, u16* __restrict__ XB) {
    int idx = blockIdx.x * 256 + threadIdx.x;       // NN*128
    int n = idx >> 7, c = idx & 127;
    float f0 = nff[n * 4], f1 = nff[n * 4 + 1], f2v = nff[n * 4 + 2], f3 = nff[n * 4 + 3];
    float v = W4[c] * f0 + W4[128 + c] * f1 + W4[256 + c] * f2v + W4[384 + c] * f3 + b4[c];
    int i0 = nif[n * 3], i1 = nif[n * 3 + 1], i2 = nif[n * 3 + 2];
    v += (tEmb[(size_t)i0 * 128 + c] + tEmb[12800 + (size_t)i1 * 128 + c]
          + tEmb[25600 + (size_t)i2 * 128 + c]) * (1.f / 3.f);
    if (!(mrand[n] < 0.5f)) v = 0.f;
    X[idx] = v; XB[idx] = f2bf(v);
}

// GINE message: m = relu(h[src] + edge_attr); atomic scatter-add into agg[dst]
// mode 0: edge_attr = 0.5*(emb0+emb1)*edge_mask (on the fly from tables)
// mode 1: edge_attr = x[src] + x[dst]
__global__ void k_message(const float* __restrict__ h, const int* __restrict__ ei,
                          const float* __restrict__ mrand, const int* __restrict__ eat,
                          const float* __restrict__ eemb, const float* __restrict__ xin,
                          int mode, float* __restrict__ agg) {
    unsigned idx = blockIdx.x * 256u + threadIdx.x; // EE*32 threads, 4 ch each
    unsigned e = idx >> 5;
    unsigned c = (idx & 31u) * 4u;
    int s = ei[e], d = ei[EE + e];
    size_t sb = (size_t)s * 128 + c, db = (size_t)d * 128 + c;
    float4 hv = *reinterpret_cast<const float4*>(h + sb);
    float ev[4];
    if (mode == 0) {
        int a0 = eat[(size_t)e * 2], a1 = eat[(size_t)e * 2 + 1];
        float em = (mrand[s] < 0.5f && mrand[d] < 0.5f) ? 0.5f : 0.f;
        float4 e0 = *reinterpret_cast<const float4*>(eemb + (size_t)a0 * 128 + c);
        float4 e1 = *reinterpret_cast<const float4*>(eemb + 12800 + (size_t)a1 * 128 + c);
        ev[0] = (e0.x + e1.x) * em; ev[1] = (e0.y + e1.y) * em;
        ev[2] = (e0.z + e1.z) * em; ev[3] = (e0.w + e1.w) * em;
    } else {
        float4 xs = *reinterpret_cast<const float4*>(xin + sb);
        float4 xd = *reinterpret_cast<const float4*>(xin + db);
        ev[0] = xs.x + xd.x; ev[1] = xs.y + xd.y; ev[2] = xs.z + xd.z; ev[3] = xs.w + xd.w;
    }
    float hv4[4] = {hv.x, hv.y, hv.z, hv.w};
#pragma unroll
    for (int j = 0; j < 4; ++j) {
        float m = hv4[j] + ev[j];
        if (m > 0.f) atomicAdd(&agg[db + j], m);    // skip zero contributions
    }
}

// elementwise: v = a (+ b); optional f32 and bf16 outputs (either may alias a/b)
__global__ void k_addcv(const float* __restrict__ a, const float* __restrict__ b,
                        float* __restrict__ outF, u16* __restrict__ outB) {
    size_t i = (size_t)blockIdx.x * 256 + threadIdx.x;
    float v = a[i] + (b ? b[i] : 0.f);
    if (outF) outF[i] = v;
    if (outB) outB[i] = f2bf(v);
}

// GraphNorm over [G, Np, C]: block=graph, thread=channel.
// Whole 128x128 tile staged in LDS (64 KB of the WGP's 320 KB) -> one global
// read instead of three. Each thread touches only its own column: no barriers.
__global__ void k_gnorm(const float* __restrict__ in1, const float* __restrict__ in2,
                        const float* __restrict__ gw, const float* __restrict__ gb,
                        const float* __restrict__ alpha,
                        float* __restrict__ dstF, u16* __restrict__ dstB) {
    __shared__ float sd[NP_ * CC];
    int g = blockIdx.x, c = threadIdx.x;
    size_t base = (size_t)g * NP_ * CC + c;
    float s = 0.f;
    for (int n = 0; n < NP_; ++n) {
        float v = in1[base + (size_t)n * CC];
        if (in2) v += in2[base + (size_t)n * CC];
        sd[n * CC + c] = v;
        s += v;
    }
    float msh = (s * (1.f / NP_)) * alpha[c];
    float s2 = 0.f;
    for (int n = 0; n < NP_; ++n) {
        float o = sd[n * CC + c] - msh;
        s2 += o * o;
    }
    float r  = rsqrtf(s2 * (1.f / NP_) + 1e-5f);
    float wc = gw[c], bc = gb[c];
    for (int n = 0; n < NP_; ++n) {
        float o = (sd[n * CC + c] - msh) * r * wc + bc;
        dstF[base + (size_t)n * CC] = o;
        if (dstB) dstB[base + (size_t)n * CC] = f2bf(o);
    }
}

// MHA core: block = (graph, head); 128 threads = query rows; dh=16
__global__ void k_attn(const float* __restrict__ qkv, u16* __restrict__ oB) {
    int g = blockIdx.x, hh = blockIdx.y, t = threadIdx.x;
    __shared__ float ks[128][17];
    __shared__ float vs[128][17];
    const float* row = qkv + ((size_t)g * NP_ + t) * 384 + hh * 16;
    float q[16];
#pragma unroll
    for (int d = 0; d < 16; ++d) {
        q[d]     = row[d];
        ks[t][d] = row[128 + d];
        vs[t][d] = row[256 + d];
    }
    __syncthreads();
    float mx = -3.4e38f;
    for (int k = 0; k < 128; ++k) {
        float s = 0.f;
#pragma unroll
        for (int d = 0; d < 16; ++d) s += q[d] * ks[k][d];
        mx = fmaxf(mx, s * 0.25f);
    }
    float sum = 0.f, o[16];
#pragma unroll
    for (int d = 0; d < 16; ++d) o[d] = 0.f;
    for (int k = 0; k < 128; ++k) {
        float s = 0.f;
#pragma unroll
        for (int d = 0; d < 16; ++d) s += q[d] * ks[k][d];
        float e = __expf(s * 0.25f - mx);
        sum += e;
#pragma unroll
        for (int d = 0; d < 16; ++d) o[d] += e * vs[k][d];
    }
    float inv = 1.f / sum;
    u16* op = oB + ((size_t)g * NP_ + t) * CC + hh * 16;
#pragma unroll
    for (int d = 0; d < 16; ++d) op[d] = f2bf(o[d] * inv);
}

// edge head input: s = tx[src] + tx[dst] (bf16) for one chunk
__global__ void k_sbuild(const float* __restrict__ tx, const int* __restrict__ ei,
                         unsigned ebase, u16* __restrict__ sb) {
    unsigned idx = blockIdx.x * 256u + threadIdx.x; // ECHUNK*32
    unsigned i = idx >> 5;
    unsigned c = (idx & 31u) * 4u;
    unsigned e = ebase + i;
    int s = ei[e], d = ei[EE + e];
    float4 a = *reinterpret_cast<const float4*>(tx + (size_t)s * 128 + c);
    float4 b = *reinterpret_cast<const float4*>(tx + (size_t)d * 128 + c);
    u16* p = sb + (size_t)i * 128 + c;
    p[0] = f2bf(a.x + b.x); p[1] = f2bf(a.y + b.y);
    p[2] = f2bf(a.z + b.z); p[3] = f2bf(a.w + b.w);
}

// cross-entropy + accuracy accumulation over [items, 2, 100] logits (ld=256)
// mode 0: node weights from mask_rand; mode 1: edge weights from endpoints
__global__ void k_ce(const float* __restrict__ logits, const int* __restrict__ labels,
                     int lstride, const float* __restrict__ mrand,
                     const int* __restrict__ ei, unsigned ebase, int mode,
                     float* __restrict__ acc) {
    unsigned idx = blockIdx.x * 256u + threadIdx.x;
    unsigned item = idx >> 1, grp = idx & 1u;
    const float* lp = logits + (size_t)item * 256 + grp * 100;
    __builtin_prefetch(lp, 0, 3);                  // global_prefetch: pull the
    __builtin_prefetch(lp + 64, 0, 3);             // 400B logit row while the
                                                   // mask/index loads resolve
    float w;
    size_t lrow;
    if (mode == 0) {
        w = (mrand[item] < 0.5f) ? 0.f : 1.f;
        lrow = (size_t)item * lstride;
    } else {
        unsigned e = ebase + item;
        int s = ei[e], d = ei[EE + e];
        w = (mrand[s] < 0.5f && mrand[d] < 0.5f) ? 0.f : 1.f;
        lrow = (size_t)(ebase + item) * lstride;
    }
    int lab = labels[lrow + grp];
    float mx = -3.4e38f; int am = 0;
    for (int j = 0; j < 100; ++j) {
        float v = lp[j];
        if (v > mx) { mx = v; am = j; }
    }
    float se = 0.f;
    for (int j = 0; j < 100; ++j) se += __expf(lp[j] - mx);
    float ce  = -((lp[lab] - mx) - __logf(se));
    float hit = (am == lab) ? 1.f : 0.f;

    __shared__ float sc[256], sh[256], sw[256];
    int t = threadIdx.x;
    sc[t] = ce * w; sh[t] = hit * w; sw[t] = w;
    __syncthreads();
    for (int st = 128; st > 0; st >>= 1) {
        if (t < st) { sc[t] += sc[t + st]; sh[t] += sh[t + st]; sw[t] += sw[t + st]; }
        __syncthreads();
    }
    if (t == 0) {
        atomicAdd(acc + 0, sc[0]);
        atomicAdd(acc + 1, sh[0]);
        atomicAdd(acc + 2, sw[0]);
    }
}

__global__ void k_fin(const float* __restrict__ acc, float* __restrict__ out) {
    float wn = fmaxf(acc[2], 1.f), we = fmaxf(acc[5], 1.f);
    out[0] = acc[0] / wn + acc[3] / we;
    out[1] = (acc[1] / wn + acc[4] / we) * 0.5f;
}

// ===========================================================================
extern "C" void kernel_launch(void* const* d_in, const int* in_sizes, int n_in,
                              void* d_out, int out_size, void* d_ws, size_t ws_size,
                              hipStream_t stream) {
    (void)in_sizes; (void)n_in; (void)out_size; (void)ws_size;
    auto F = [&](int i) { return (const float*)d_in[i]; };
    auto I = [&](int i) { return (const int*)d_in[i]; };

    // d_in layout (jax pytree: dicts sorted by key, lists in order):
    // 0 node_int_feats, 1 node_float_feats, 2 edge_attrs, 3 edge_index, 4 mask_rand,
    // 5 edge_emb, 6 edge_recovery.W, 7 edge_recovery.b,
    // 8.. layers[0] (28 leaves), 36.. layers[1],
    // 64 node_float.W, 65 node_float.b, 66 node_int_emb, 67 recovery.W, 68 recovery.b
    const int LB[2] = {8, 36};
    const int*   nif   = I(0);
    const float* nff   = F(1);
    const int*   eat   = I(2);
    const int*   ei    = I(3);
    const float* mrand = F(4);

    // -------- workspace layout --------
    char* ws = (char*)d_ws;
    float* X0  = (float*)ws;  ws += (size_t)NN * 128 * 4;   // layer-0 x (f32)
    float* H   = (float*)ws;  ws += (size_t)NN * 128 * 4;   // gine h / h1
    float* R2  = (float*)ws;  ws += (size_t)NN * 128 * 4;   // attn+res / out / mlp
    float* AGG = (float*)ws;  ws += (size_t)NN * 128 * 4;   // scatter accumulator
    float* QKV = (float*)ws;  ws += (size_t)NN * 384 * 4;   // in_proj output
    float* LOG = (float*)ws;  ws += (size_t)NN * 256 * 4;   // head logits (padded)
    u16* X0B = (u16*)ws;      ws += (size_t)NN * 128 * 2;   // bf16 of layer-0 x
    u16* XB  = (u16*)ws;      ws += (size_t)NN * 128 * 2;   // bf16 of layer output x
    u16* UB  = (u16*)ws;      ws += (size_t)NN * 128 * 2;   // bf16 temp (gine/attn/s)
    u16* T1B = (u16*)ws;      ws += (size_t)NN * 256 * 2;   // bf16 hidden (<=256 wide)
    u16* WBF = (u16*)ws;      ws += (size_t)458752 * 2;     // bf16 transposed weights
    float* ACC = (float*)ws;                                // 6 loss/acc accumulators

    float* OUT = (float*)d_out;                             // [2,N,C] ++ loss ++ acc

    // -------- weight prep: transpose + bf16 --------
    size_t wo = 0;
    auto prep = [&](int leafW, int K, int Nout, int NP) -> size_t {
        size_t off = wo;
        int tot = NP * K;
        k_prep_w<<<tot / 256, 256, 0, stream>>>(F(leafW), WBF + off, K, Nout, NP);
        wo += (size_t)tot;
        return off;
    };
    size_t wInP[2], wOutP[2], wG[2][2][2], wM1[2], wM2[2];
    for (int l = 0; l < 2; ++l) {
        wInP[l]     = prep(LB[l] + 0, 128, 384, 384);
        wOutP[l]    = prep(LB[l] + 2, 128, 128, 128);
        wG[l][0][0] = prep(LB[l] + 5, 128, 128, 128);
        wG[l][0][1] = prep(LB[l] + 7, 128, 128, 128);
        wG[l][1][0] = prep(LB[l] + 10, 128, 128, 128);
        wG[l][1][1] = prep(LB[l] + 12, 128, 128, 128);
        wM1[l]      = prep(LB[l] + 15, 128, 256, 256);
        wM2[l]      = prep(LB[l] + 17, 256, 128, 128);
    }
    size_t wRec  = prep(67, 128, 200, 256);
    size_t wERec = prep(6,  128, 200, 256);

    hipMemsetAsync(ACC, 0, 8 * sizeof(float), stream);

    // -------- embedding --------
    k_embed<<<NN * 128 / 256, 256, 0, stream>>>(nif, nff, mrand, F(64), F(65), F(66), X0, X0B);

    const float* xF = X0;
    const u16*   xB = X0B;

    dim3 g32(NN / 128, 2);   // GEMM grids: block tile 128x64
    dim3 g64(NN / 128, 4);
    dim3 g96(NN / 128, 6);

    for (int l = 0; l < 2; ++l) {
        const float* a1[2] = {F(LB[l] + 4), F(LB[l] + 9)};
        const float* l1b[2] = {F(LB[l] + 6), F(LB[l] + 11)};
        const float* l2b[2] = {F(LB[l] + 8), F(LB[l] + 13)};
        const size_t wl1[2] = {wG[l][0][0], wG[l][1][0]};
        const size_t wl2[2] = {wG[l][0][1], wG[l][1][1]};

        // ---- GINE x2 ----
        const float* hRead = xF;
        for (int it = 0; it < 2; ++it) {
            hipMemsetAsync(AGG, 0, (size_t)NN * 128 * 4, stream);
            k_message<<<EE * 32 / 256, 256, 0, stream>>>(hRead, ei, mrand, eat, F(5),
                                                         xF, (l == 0) ? 0 : 1, AGG);
            k_addcv<<<NN * 128 / 256, 256, 0, stream>>>(hRead, AGG, nullptr, UB);
            k_gemm<<<g32, 256, 0, stream>>>(UB, WBF + wl1[it], l1b[it], a1[it], nullptr,
                                            nullptr, T1B, NN, 128, 128, 128);
            k_gemm<<<g32, 256, 0, stream>>>(T1B, WBF + wl2[it], l2b[it], nullptr, nullptr,
                                            H, nullptr, NN, 128, 128, 128);
            hRead = H;
        }
        // h1 = graphnorm1(h + x)  -> H
        k_gnorm<<<GG, 128, 0, stream>>>(H, xF, F(LB[l] + 21), F(LB[l] + 19), F(LB[l] + 20),
                                        H, nullptr);
        // ---- MHA branch ----
        k_gemm<<<g96, 256, 0, stream>>>(xB, WBF + wInP[l], F(LB[l] + 1), nullptr, nullptr,
                                        QKV, nullptr, NN, 128, 384, 384);
        k_attn<<<dim3(GG, 8), 128, 0, stream>>>(QKV, UB);
        k_gemm<<<g32, 256, 0, stream>>>(UB, WBF + wOutP[l], F(LB[l] + 3), nullptr, xF,
                                        R2, nullptr, NN, 128, 128, 128);
        // h2 = graphnorm2(attn + x) -> R2 (in-place safe: LDS-staged)
        k_gnorm<<<GG, 128, 0, stream>>>(R2, nullptr, F(LB[l] + 24), F(LB[l] + 22),
                                        F(LB[l] + 23), R2, nullptr);
        // out = h1 + h2 -> R2 (f32) + UB (bf16)
        k_addcv<<<NN * 128 / 256, 256, 0, stream>>>(H, R2, R2, UB);
        // ---- MLP with residual ----
        k_gemm<<<g64, 256, 0, stream>>>(UB, WBF + wM1[l], F(LB[l] + 16), F(LB[l] + 14),
                                        nullptr, nullptr, T1B, NN, 128, 256, 256);
        k_gemm<<<g32, 256, 0, stream>>>(T1B, WBF + wM2[l], F(LB[l] + 18), nullptr, R2,
                                        R2, nullptr, NN, 256, 128, 128);
        // x_next = graphnorm3(R2) -> d_out slice l (f32) + XB (bf16)
        float* xNext = OUT + (size_t)l * NN * CC;
        k_gnorm<<<GG, 128, 0, stream>>>(R2, nullptr, F(LB[l] + 27), F(LB[l] + 25),
                                        F(LB[l] + 26), xNext, XB);
        xF = xNext; xB = XB;
    }

    // -------- node recovery head --------
    k_gemm<<<g64, 256, 0, stream>>>(XB, WBF + wRec, F(68), nullptr, nullptr,
                                    LOG, nullptr, NN, 128, 256, 200);
    k_ce<<<NN * 2 / 256, 256, 0, stream>>>(LOG, nif, 3, mrand, ei, 0, 0, ACC);

    // -------- edge recovery head, chunked (reuses UB + LOG) --------
    const float* tx = OUT + (size_t)NN * CC;   // final layer output
    dim3 gE(ECHUNK / 128, 4);
    for (unsigned cb = 0; cb < EE; cb += ECHUNK) {
        k_sbuild<<<ECHUNK * 32 / 256, 256, 0, stream>>>(tx, ei, cb, UB);
        k_gemm<<<gE, 256, 0, stream>>>(UB, WBF + wERec, F(7), nullptr, nullptr,
                                       LOG, nullptr, ECHUNK, 128, 256, 200);
        k_ce<<<ECHUNK * 2 / 256, 256, 0, stream>>>(LOG, eat, 2, mrand, ei, cb, 1, ACC + 3);
    }

    k_fin<<<1, 1, 0, stream>>>(ACC, OUT + (size_t)2 * NN * CC);
}
